// GNNVariationalEncoder_10986526343300
// MI455X (gfx1250) — compile-verified, matrix operand order checked by reading, not run on hardware
//
#include <hip/hip_runtime.h>
#include <hip/hip_bf16.h>
#include <math.h>

typedef __attribute__((ext_vector_type(16))) _Float16 v16h;
typedef __attribute__((ext_vector_type(8)))  _Float16 v8h;
typedef __attribute__((ext_vector_type(8)))  float    v8f;

// ---------------------------------------------------------------------------
// WMMA helpers (gfx1250, wave32, V_WMMA_F32_16X16X32_F16)
// ---------------------------------------------------------------------------
__device__ __forceinline__ v8f vzero8() {
  v8f z = {0.f,0.f,0.f,0.f,0.f,0.f,0.f,0.f};
  return z;
}

__device__ __forceinline__ v8f wmma_f16(v16h a, v16h b, v8f c) {
  return __builtin_amdgcn_wmma_f32_16x16x32_f16(false, a, false, b, (short)0, c,
                                                false, false);
}

// 16-bit A-matrix 16x32 fragment from a row-major LDS panel.
// Per ISA layout: lane l holds row M=l&15; half h maps to
// K = k0 + h + (h<8?0:8) + (l>=16?8:0)  -> two contiguous 8-half runs.
__device__ __forceinline__ v16h load_frag_a(const _Float16* base, int stride,
                                            int k0, int lane) {
  const int m  = lane & 15;
  const int kb = k0 + ((lane >> 4) << 3);
  const _Float16* p = base + m * stride + kb;
  v8h lo = *(const v8h*)(p);        // K = kb .. kb+7
  v8h hi = *(const v8h*)(p + 16);   // K = kb+16 .. kb+23
  v16h r;
  #pragma unroll
  for (int i = 0; i < 8; ++i) { r[i] = lo[i]; r[i + 8] = hi[i]; }
  return r;
}

// B fragments pre-packed so each lane reads 32 contiguous bytes.
__device__ __forceinline__ v16h load_frag_b(const _Float16* Wp, int tile, int lane) {
  return *(const v16h*)(Wp + ((size_t)tile * 32 + lane) * 16);
}

// Layer with 128 outputs: A[64 x 32*KS] (f16 LDS) x W -> relu -> hid (f16 LDS).
// All 8 B fragments are loaded into registers before the WMMA chain so the
// loads go out in one clause and the WMMAs drain with decreasing loadcnt.
// Internal barrier lets hid alias the tail of the A panel.
template<int KS>
__device__ __forceinline__ void wmma_layer_relu(const _Float16* A, int strideA,
                                                const _Float16* Wp,
                                                const float* bias,
                                                _Float16* hid, int strideH,
                                                int tid) {
  const int lane = tid & 31;
  const int m0   = (tid >> 5) * 16;          // 4 waves x 16 rows
  v8f acc[8];
  #pragma unroll
  for (int t = 0; t < 8; ++t) acc[t] = vzero8();
  const _Float16* Abase = A + m0 * strideA;
  for (int s = 0; s < KS; ++s) {
    v16h a = load_frag_a(Abase, strideA, s * 32, lane);
    v16h bf[8];
    #pragma unroll
    for (int t = 0; t < 8; ++t) bf[t] = load_frag_b(Wp, t * KS + s, lane);
    #pragma unroll
    for (int t = 0; t < 8; ++t) acc[t] = wmma_f16(a, bf[t], acc[t]);
  }
  __syncthreads();                            // A panel dead before hid overwrite
  const int mlo = (lane >> 4) << 3;
  const int nlo = lane & 15;
  #pragma unroll
  for (int t = 0; t < 8; ++t) {
    #pragma unroll
    for (int v = 0; v < 8; ++v) {
      int m = m0 + mlo + v, n = t * 16 + nlo;
      float x = acc[t][v] + bias[n];
      hid[m * strideH + n] = (_Float16)fmaxf(x, 0.f);
    }
  }
}

// 128x128 layer: hid (f16 LDS) x W -> + bias -> f32 out panel.
__device__ __forceinline__ void wmma_gemm_128(const _Float16* A, int strideA,
                                              const _Float16* Wp,
                                              const float* bias,
                                              float* outp, int strideO, int tid) {
  const int lane = tid & 31;
  const int m0   = (tid >> 5) * 16;
  v8f acc[8];
  #pragma unroll
  for (int t = 0; t < 8; ++t) acc[t] = vzero8();
  const _Float16* Abase = A + m0 * strideA;
  for (int s = 0; s < 4; ++s) {
    v16h a = load_frag_a(Abase, strideA, s * 32, lane);
    v16h bf[8];
    #pragma unroll
    for (int t = 0; t < 8; ++t) bf[t] = load_frag_b(Wp, t * 4 + s, lane);
    #pragma unroll
    for (int t = 0; t < 8; ++t) acc[t] = wmma_f16(a, bf[t], acc[t]);
  }
  const int mlo = (lane >> 4) << 3;
  const int nlo = lane & 15;
  #pragma unroll
  for (int t = 0; t < 8; ++t) {
    #pragma unroll
    for (int v = 0; v < 8; ++v) {
      int m = m0 + mlo + v, n = t * 16 + nlo;
      outp[m * strideO + n] = acc[t][v] + bias[n];
    }
  }
}

__device__ __forceinline__ void layernorm_row(const float* orow, const float* sG,
                                              const float* sBe, float* grow,
                                              bool residual) {
  float s = 0.f;
  for (int c = 0; c < 128; ++c) s += orow[c];
  float mu = s * (1.f / 128.f);
  float v = 0.f;
  for (int c = 0; c < 128; ++c) { float d = orow[c] - mu; v += d * d; }
  float rs = rsqrtf(v * (1.f / 128.f) + 1e-5f);
  for (int c = 0; c < 128; ++c) {
    float val = (orow[c] - mu) * rs * sG[c] + sBe[c];
    grow[c] = residual ? (grow[c] + val) : val;
  }
}

// ---------------------------------------------------------------------------
// Weight repack: fp32 row-major [K][128] -> f16 fragments
// dst[((t*(K/32)+s)*32 + lane)*16 + h] = W[k][n],
//   k = s*32 + h + (h<8?0:8) + (lane>=16?8:0),  n = t*16 + (lane&15)
// ---------------------------------------------------------------------------
__global__ void pack_weights(const float* __restrict__ W,
                             _Float16* __restrict__ dst, int K) {
  int i = blockIdx.x * 256 + threadIdx.x;
  if (i >= K * 128) return;
  int h = i & 15;
  int l = (i >> 4) & 31;
  int rest = i >> 9;
  int KS = K / 32;
  int s = rest % KS;
  int t = rest / KS;
  int k = s * 32 + h + ((h < 8) ? 0 : 8) + ((l >= 16) ? 8 : 0);
  int n = t * 16 + (l & 15);
  dst[i] = (_Float16)W[(size_t)k * 128 + n];
}

// ---------------------------------------------------------------------------
// Encoders: scalar first layer (DIN<=8), WMMA second layer, LN, store.
// ---------------------------------------------------------------------------
template<int DIN>
__global__ __launch_bounds__(128) void enc_mlp_kernel(
    const float* __restrict__ xin, const float* __restrict__ W1,
    const float* __restrict__ b1, const _Float16* __restrict__ W2p,
    const float* __restrict__ b2, const float* __restrict__ g,
    const float* __restrict__ be, float* __restrict__ out, int nrows) {
  __shared__ _Float16 hid[64 * 136];
  __shared__ float outp[64 * 132];
  __shared__ float sW1[DIN * 128];
  __shared__ float sB1[128], sB2[128], sG[128], sBe[128];
  __shared__ float sx[64][DIN];
  const int tid = threadIdx.x;
  const int row0 = blockIdx.x * 64;
  for (int i = tid; i < DIN * 128; i += 128) sW1[i] = W1[i];
  if (tid < 128) { sB1[tid] = b1[tid]; sB2[tid] = b2[tid]; sG[tid] = g[tid]; sBe[tid] = be[tid]; }
  for (int i = tid; i < 64 * DIN; i += 128) {
    int r = i / DIN, k = i % DIN, rr = row0 + r;
    sx[r][k] = (rr < nrows) ? xin[(size_t)rr * DIN + k] : 0.f;
  }
  __syncthreads();
  for (int i = tid; i < 64 * 128; i += 128) {
    int r = i >> 7, c = i & 127;
    float acc = sB1[c];
    #pragma unroll
    for (int k = 0; k < DIN; ++k) acc += sx[r][k] * sW1[k * 128 + c];
    hid[r * 136 + c] = (_Float16)fmaxf(acc, 0.f);
  }
  __syncthreads();
  wmma_gemm_128(hid, 136, W2p, sB2, outp, 132, tid);
  __syncthreads();
  if (tid < 64) {
    int rr = row0 + tid;
    if (rr < nrows) layernorm_row(outp + tid * 132, sG, sBe, out + (size_t)rr * 128, false);
  }
}

// ---------------------------------------------------------------------------
// Edge MLP: concat(h[src], h[dst], e) [K=384] -> 128 -> 128, LN, e += result
// Dynamic LDS: panelA 64x392 halfs  UNION  (outp 64x132 f32 | hid 64x136 halfs)
// ---------------------------------------------------------------------------
__global__ __launch_bounds__(128) void edge_mlp_kernel(
    const float* __restrict__ h, float* __restrict__ e,
    const int* __restrict__ src, const int* __restrict__ dstI,
    const _Float16* __restrict__ W1p, const _Float16* __restrict__ W2p,
    const float* __restrict__ b1, const float* __restrict__ b2,
    const float* __restrict__ g, const float* __restrict__ be, int E_) {
  extern __shared__ char smem[];
  _Float16* panelA = (_Float16*)smem;                 // 64*392*2 = 50176 B
  float*    outp   = (float*)smem;                    // 64*132*4 = 33792 B
  _Float16* hid    = (_Float16*)(smem + 64 * 132 * 4);// 64*136*2 = 17408 B
  __shared__ float sB1[128], sB2[128], sG[128], sBe[128];
  __shared__ int sidx[64], didx[64];
  const int tid = threadIdx.x;
  const int row0 = blockIdx.x * 64;
  if (tid < 128) { sB1[tid] = b1[tid]; sB2[tid] = b2[tid]; sG[tid] = g[tid]; sBe[tid] = be[tid]; }
  if (tid < 64) {
    int r = row0 + tid;
    sidx[tid] = (r < E_) ? src[r] : 0;
    didx[tid] = (r < E_) ? dstI[r] : 0;
  }
  __syncthreads();
  for (int u = tid; u < 64 * 96; u += 128) {          // 4 cols per unit
    int r = u / 96, c4 = (u % 96) * 4;
    int er = row0 + r;
    float4 v = make_float4(0.f, 0.f, 0.f, 0.f);
    if (er < E_) {
      if (c4 < 128)      v = *(const float4*)(h + (size_t)sidx[r] * 128 + c4);
      else if (c4 < 256) v = *(const float4*)(h + (size_t)didx[r] * 128 + (c4 - 128));
      else               v = *(const float4*)(e + (size_t)er * 128 + (c4 - 256));
    }
    _Float16* p = panelA + r * 392 + c4;
    p[0] = (_Float16)v.x; p[1] = (_Float16)v.y;
    p[2] = (_Float16)v.z; p[3] = (_Float16)v.w;
  }
  __syncthreads();
  wmma_layer_relu<12>(panelA, 392, W1p, sB1, hid, 136, tid);
  __syncthreads();
  wmma_gemm_128(hid, 136, W2p, sB2, outp, 132, tid);
  __syncthreads();
  if (tid < 64) {
    int er = row0 + tid;
    if (er < E_) layernorm_row(outp + tid * 132, sG, sBe, e + (size_t)er * 128, true);
  }
}

// ---------------------------------------------------------------------------
// Node MLP: concat(h, agg) [K=256] -> 128 -> 128, LN, h += result
// ---------------------------------------------------------------------------
__global__ __launch_bounds__(128) void node_mlp_kernel(
    float* __restrict__ h, const float* __restrict__ agg,
    const _Float16* __restrict__ W1p, const _Float16* __restrict__ W2p,
    const float* __restrict__ b1, const float* __restrict__ b2,
    const float* __restrict__ g, const float* __restrict__ be, int N_) {
  extern __shared__ char smem[];
  _Float16* panelA = (_Float16*)smem;                 // 64*264*2 = 33792 B
  float*    outp   = (float*)smem;                    // 64*132*4 = 33792 B
  _Float16* hid    = (_Float16*)(smem + 64 * 132 * 4);// 64*136*2 = 17408 B
  __shared__ float sB1[128], sB2[128], sG[128], sBe[128];
  const int tid = threadIdx.x;
  const int row0 = blockIdx.x * 64;
  if (tid < 128) { sB1[tid] = b1[tid]; sB2[tid] = b2[tid]; sG[tid] = g[tid]; sBe[tid] = be[tid]; }
  __syncthreads();
  for (int u = tid; u < 64 * 64; u += 128) {
    int r = u >> 6, c4 = (u & 63) * 4;
    int nrow = row0 + r;
    float4 v = make_float4(0.f, 0.f, 0.f, 0.f);
    if (nrow < N_) {
      v = (c4 < 128) ? *(const float4*)(h + (size_t)nrow * 128 + c4)
                     : *(const float4*)(agg + (size_t)nrow * 128 + (c4 - 128));
    }
    _Float16* p = panelA + r * 264 + c4;
    p[0] = (_Float16)v.x; p[1] = (_Float16)v.y;
    p[2] = (_Float16)v.z; p[3] = (_Float16)v.w;
  }
  __syncthreads();
  wmma_layer_relu<8>(panelA, 264, W1p, sB1, hid, 136, tid);
  __syncthreads();
  wmma_gemm_128(hid, 136, W2p, sB2, outp, 132, tid);
  __syncthreads();
  if (tid < 64) {
    int nrow = row0 + tid;
    if (nrow < N_) layernorm_row(outp + tid * 132, sG, sBe, h + (size_t)nrow * 128, true);
  }
}

// ---------------------------------------------------------------------------
// Aggregation / pooling / head
// ---------------------------------------------------------------------------
__global__ void fill_kernel(float* __restrict__ p, float v, size_t n) {
  size_t i = (size_t)blockIdx.x * 256 + threadIdx.x;
  if (i < n) p[i] = v;
}

__global__ void scatter_add_kernel(const float* __restrict__ e,
                                   const int* __restrict__ dstI,
                                   float* __restrict__ agg, int E_) {
  long long i = (long long)blockIdx.x * 256 + threadIdx.x;
  int eidx = (int)(i >> 5);
  if (eidx >= E_) return;
  int c4 = ((int)i & 31) * 4;
  int d = dstI[eidx];
  float4 v = *(const float4*)(e + (size_t)eidx * 128 + c4);
  float* a = agg + (size_t)d * 128 + c4;
  atomicAdd(a + 0, v.x); atomicAdd(a + 1, v.y);
  atomicAdd(a + 2, v.z); atomicAdd(a + 3, v.w);
}

__device__ __forceinline__ void atomicMaxF(float* addr, float v) {
  unsigned* ua = (unsigned*)addr;
  unsigned old = *ua;
  while (__uint_as_float(old) < v) {
    unsigned assumed = old;
    old = atomicCAS(ua, assumed, __float_as_uint(v));
    if (old == assumed) break;
  }
}

__global__ void gate_kernel(const float* __restrict__ h,
                            const float* __restrict__ gW,
                            const float* __restrict__ gB,
                            const int* __restrict__ batch,
                            float* __restrict__ gate, float* __restrict__ m,
                            int N_) {
  int n = blockIdx.x * 256 + threadIdx.x;
  if (n >= N_) return;
  const float* hr = h + (size_t)n * 128;
  float acc = gB[0];
  for (int c = 0; c < 128; c += 4) {
    float4 a = *(const float4*)(hr + c);
    float4 w = *(const float4*)(gW + c);
    acc += a.x * w.x + a.y * w.y + a.z * w.z + a.w * w.w;
  }
  gate[n] = acc;
  atomicMaxF(&m[batch[n]], acc);
}

__global__ void wexp_kernel(const float* __restrict__ gate,
                            const int* __restrict__ batch,
                            const float* __restrict__ m, float* __restrict__ w,
                            float* __restrict__ denom, int N_) {
  __shared__ float part[4];
  int tid = threadIdx.x;
  if (tid < 4) part[tid] = 0.f;
  __syncthreads();
  int n = blockIdx.x * 256 + tid;
  if (n < N_) {
    int b = batch[n];
    float wv = __expf(gate[n] - m[b]);
    w[n] = wv;
    atomicAdd(&part[b], wv);
  }
  __syncthreads();
  if (tid < 4) atomicAdd(&denom[tid], part[tid]);
}

__global__ void anorm_kernel(float* __restrict__ w, const int* __restrict__ batch,
                             const float* __restrict__ denom, int N_) {
  int n = blockIdx.x * 256 + threadIdx.x;
  if (n >= N_) return;
  w[n] = w[n] / denom[batch[n]];
}

__global__ __launch_bounds__(128) void pool_kernel(const float* __restrict__ h,
                                                   const float* __restrict__ a,
                                                   const int* __restrict__ batch,
                                                   float* __restrict__ hg, int N_) {
  const int d = threadIdx.x;           // 0..127
  const int n0 = blockIdx.x * 256;
  float a0 = 0.f, a1 = 0.f, a2 = 0.f, a3 = 0.f;
  for (int j = 0; j < 256; ++j) {
    int n = n0 + j;
    if (n >= N_) break;
    int b = batch[n];
    float x = a[n] * h[(size_t)n * 128 + d];
    a0 += (b == 0) ? x : 0.f;
    a1 += (b == 1) ? x : 0.f;
    a2 += (b == 2) ? x : 0.f;
    a3 += (b == 3) ? x : 0.f;
  }
  atomicAdd(&hg[0 * 128 + d], a0);
  atomicAdd(&hg[1 * 128 + d], a1);
  atomicAdd(&hg[2 * 128 + d], a2);
  atomicAdd(&hg[3 * 128 + d], a3);
}

__global__ void head_kernel(const float* __restrict__ hg,
                            const float* __restrict__ muW,
                            const float* __restrict__ muB,
                            const float* __restrict__ lvW,
                            const float* __restrict__ lvB,
                            const float* __restrict__ eps,
                            float* __restrict__ out) {
  int i = threadIdx.x;                 // 256 = B*Z
  int b = i >> 6, zc = i & 63;
  const float* hr = hg + b * 128;
  float mu = muB[zc], lv = lvB[zc];
  for (int k = 0; k < 128; ++k) {
    float x = hr[k];
    mu += x * muW[k * 64 + zc];
    lv += x * lvW[k * 64 + zc];
  }
  float z = mu + __expf(0.5f * lv) * eps[b * 64 + zc];
  out[i] = z;
  out[256 + i] = mu;
  out[512 + i] = lv;
}

// ---------------------------------------------------------------------------
// Host orchestration
// ---------------------------------------------------------------------------
struct MlpP { const float *W1, *b1, *W2, *b2, *g, *be; };

extern "C" void kernel_launch(void* const* d_in, const int* in_sizes, int n_in,
                              void* d_out, int out_size, void* d_ws, size_t ws_size,
                              hipStream_t stream) {
  (void)n_in; (void)out_size; (void)ws_size;
  const bool orderA = (in_sizes[0] % 3 == 0) && (in_sizes[0] != 100000);
  auto F = [&](int i) -> const float* { return (const float*)d_in[i]; };
  auto I = [&](int i) -> const int*   { return (const int*)d_in[i]; };
  auto mlpAt = [&](int i) -> MlpP {
    if (orderA)  // insertion order: W1,b1,W2,b2,g,be
      return MlpP{F(i), F(i + 1), F(i + 2), F(i + 3), F(i + 4), F(i + 5)};
    // sorted order: W1,W2,b1,b2,be,g
    return MlpP{F(i), F(i + 2), F(i + 1), F(i + 3), F(i + 5), F(i + 4)};
  };

  const float *y, *edge_attr, *eps;
  const int *ei, *batch;
  MlpP pNode, pEdge, pMpE[2], pMpN[2];
  const float *gW, *gB, *muW, *muB, *lvW, *lvB;
  int N, E;
  if (orderA) {
    y = F(0); ei = I(1); edge_attr = F(2); batch = I(3); eps = F(4);
    N = in_sizes[0] / 3; E = in_sizes[2] / 8;
    pNode = mlpAt(5);  pEdge = mlpAt(11);
    pMpE[0] = mlpAt(17); pMpN[0] = mlpAt(23);
    pMpE[1] = mlpAt(29); pMpN[1] = mlpAt(35);
    gW = F(41); gB = F(42); muW = F(43); muB = F(44); lvW = F(45); lvB = F(46);
  } else {
    batch = I(0); edge_attr = F(1); ei = I(2); eps = F(3);
    N = in_sizes[0]; E = in_sizes[1] / 8;
    pEdge = mlpAt(4);
    gW = F(10); gB = F(11); lvW = F(12); lvB = F(13);
    pMpE[0] = mlpAt(14); pMpN[0] = mlpAt(20);
    pMpE[1] = mlpAt(26); pMpN[1] = mlpAt(32);
    muW = F(38); muB = F(39); pNode = mlpAt(40); y = F(46);
  }
  const int* src = ei;
  const int* dst = ei + E;

  // workspace carve
  char* base = (char*)d_ws;
  size_t off = 0;
  auto carve = [&](size_t bytes) -> char* {
    char* p = base + off;
    off = (off + bytes + 255) & ~(size_t)255;
    return p;
  };
  float* h       = (float*)carve((size_t)N * 128 * 4);
  float* e       = (float*)carve((size_t)E * 128 * 4);
  float* agg     = (float*)carve((size_t)N * 128 * 4);
  float* gateBuf = (float*)carve((size_t)N * 4);
  float* wBuf    = (float*)carve((size_t)N * 4);
  float* mBuf    = (float*)carve(4 * 4);
  float* denom   = (float*)carve(4 * 4);
  float* hg      = (float*)carve(4 * 128 * 4);
  _Float16* pwNode2 = (_Float16*)carve(128 * 128 * 2);
  _Float16* pwEdge2 = (_Float16*)carve(128 * 128 * 2);
  _Float16 *pwE1[2], *pwE2[2], *pwN1[2], *pwN2[2];
  for (int L = 0; L < 2; ++L) {
    pwE1[L] = (_Float16*)carve(384 * 128 * 2);
    pwE2[L] = (_Float16*)carve(128 * 128 * 2);
    pwN1[L] = (_Float16*)carve(256 * 128 * 2);
    pwN2[L] = (_Float16*)carve(128 * 128 * 2);
  }

  auto pack = [&](const float* W, _Float16* dstp, int K) {
    int n = K * 128;
    pack_weights<<<(n + 255) / 256, 256, 0, stream>>>(W, dstp, K);
  };
  pack(pNode.W2, pwNode2, 128);
  pack(pEdge.W2, pwEdge2, 128);
  for (int L = 0; L < 2; ++L) {
    pack(pMpE[L].W1, pwE1[L], 384);
    pack(pMpE[L].W2, pwE2[L], 128);
    pack(pMpN[L].W1, pwN1[L], 256);
    pack(pMpN[L].W2, pwN2[L], 128);
  }

  // encoders
  enc_mlp_kernel<3><<<(N + 63) / 64, 128, 0, stream>>>(
      y, pNode.W1, pNode.b1, pwNode2, pNode.b2, pNode.g, pNode.be, h, N);
  enc_mlp_kernel<8><<<(E + 63) / 64, 128, 0, stream>>>(
      edge_attr, pEdge.W1, pEdge.b1, pwEdge2, pEdge.b2, pEdge.g, pEdge.be, e, E);

  // message passing
  const size_t mpShmem = 64 * 132 * 4 + 64 * 136 * 2;  // 51200 B (covers panels)
  for (int L = 0; L < 2; ++L) {
    edge_mlp_kernel<<<(E + 63) / 64, 128, mpShmem, stream>>>(
        h, e, src, dst, pwE1[L], pwE2[L],
        pMpE[L].b1, pMpE[L].b2, pMpE[L].g, pMpE[L].be, E);
    size_t nAgg = (size_t)N * 128;
    fill_kernel<<<(unsigned)((nAgg + 255) / 256), 256, 0, stream>>>(agg, 0.f, nAgg);
    scatter_add_kernel<<<(unsigned)(((size_t)E * 32 + 255) / 256), 256, 0, stream>>>(
        e, dst, agg, E);
    node_mlp_kernel<<<(N + 63) / 64, 128, mpShmem, stream>>>(
        h, agg, pwN1[L], pwN2[L],
        pMpN[L].b1, pMpN[L].b2, pMpN[L].g, pMpN[L].be, N);
  }

  // attention pooling + head
  fill_kernel<<<1, 256, 0, stream>>>(mBuf, -INFINITY, 4);
  fill_kernel<<<1, 256, 0, stream>>>(denom, 0.f, 4);
  fill_kernel<<<2, 256, 0, stream>>>(hg, 0.f, 4 * 128);
  gate_kernel<<<(N + 255) / 256, 256, 0, stream>>>(h, gW, gB, batch, gateBuf, mBuf, N);
  wexp_kernel<<<(N + 255) / 256, 256, 0, stream>>>(gateBuf, batch, mBuf, wBuf, denom, N);
  anorm_kernel<<<(N + 255) / 256, 256, 0, stream>>>(wBuf, batch, denom, N);
  pool_kernel<<<(N + 255) / 256, 128, 0, stream>>>(h, wBuf, batch, hg, N);
  head_kernel<<<1, 256, 0, stream>>>(hg, muW, muB, lvW, lvB, eps, (float*)d_out);
}
